// _HFGlmDsaAttention_62895501082713
// MI455X (gfx1250) — compile-verified, matrix operand order checked by reading, not run on hardware
//
#include <hip/hip_runtime.h>
#include <hip/hip_bf16.h>

typedef __attribute__((ext_vector_type(16))) _Float16 v16h;
typedef __attribute__((ext_vector_type(8)))  _Float16 v8h;
typedef __attribute__((ext_vector_type(8)))  float    v8f;

#define S_LEN   2048
#define HID     2048
#define NH      16
#define D_NOPE  128
#define D_ROPE  64
#define D_QK    192      // NOPE + ROPE
#define D_V     128
#define KV_LORA 512
#define Q_LORA  1536
#define IDXH    16
#define IDXD    128
#define TOPK    512
#define NEGBIG  (-1.0e30f)

// ln(10000)/32 : inv_freq(i) = 10000^(-2i/64) = exp(-i * LN1E4_32)
#define LN1E4_32 0.2878231366242557f

__device__ __forceinline__ float inv_freq(int i) {
  return __expf(-LN1E4_32 * (float)i);
}

// ---------------------------------------------------------------------------
// WMMA helpers (CDNA5 gfx1250, wave32)
// ---------------------------------------------------------------------------
__device__ __forceinline__ v8f wmma32(v16h a, v16h b, v8f c) {
  // D = A(16x32 f16) x B(32x16 f16) + C(16x16 f32)
  return __builtin_amdgcn_wmma_f32_16x16x32_f16(false, a, false, b,
                                                (short)0, c, false, false);
}

// Load a 16x32 f16 operand fragment from row-major storage (ld in halves).
// Lane r=lane&15 owns row r; kb=8*(lane>>4): halves[0..7]=K[kb..kb+7],
// halves[8..15]=K[16+kb..16+kb+7]  (ISA 7.12.2, 16-bit A 16x32 layout).
__device__ __forceinline__ v16h load_frag(const _Float16* p, int ld) {
  const int lane = threadIdx.x & 31;
  const int r    = lane & 15;
  const int kb   = (lane >> 4) << 3;
  const _Float16* b = p + r * ld;
  v8h lo = *(const v8h*)(b + kb);
  v8h hi = *(const v8h*)(b + kb + 16);
  v16h f;
#pragma unroll
  for (int i = 0; i < 8; ++i) { f[i] = lo[i]; f[8 + i] = hi[i]; }
  return f;
}

// ---------------------------------------------------------------------------
// f32 -> f16 conversion
// ---------------------------------------------------------------------------
__global__ void k_cvt(const float* __restrict__ s, _Float16* __restrict__ d, int n) {
  int i = blockIdx.x * 256 + threadIdx.x;
  if (i < n) d[i] = (_Float16)s[i];
}

// ---------------------------------------------------------------------------
// C[M,N] (f32) = A[M,K] (f16, row-major) @ B[N,K]^T (f16, row-major)
// Requires M%64==0, K%32==0.  N may be ragged (guarded).
// Block 256 threads = 8 waves; block tile 64x128; wave tile 32x32 (2x2 WMMA).
// Double-buffered LDS: next K-tile's global loads are issued before the
// current tile's WMMAs, so the s_wait_loadcnt lands behind real work and
// only ONE workgroup barrier is needed per K step.  L2 prefetch two steps
// ahead (global_prefetch_b8).
// ---------------------------------------------------------------------------
#define GTM 64
#define GTN 128
#define GTK 32
#define APITCH 40
#define BPITCH 40

__global__ __launch_bounds__(256)
void k_gemm_nt(const _Float16* __restrict__ A, const _Float16* __restrict__ B,
               float* __restrict__ C, int M, int N, int K) {
  __shared__ _Float16 As[2][GTM * APITCH];
  __shared__ _Float16 Bs[2][GTN * BPITCH];
  const int tid  = threadIdx.x;
  const int lane = tid & 31;
  const int wave = tid >> 5;
  const int wm   = wave >> 2;      // 0..1
  const int wn   = wave & 3;       // 0..3
  const int m0   = blockIdx.y * GTM;
  const int n0   = blockIdx.x * GTN;

  // per-thread staging coordinates
  const int ra = tid >> 2;            // A row 0..63
  const int ca = (tid & 3) << 3;      // A col group
  const _Float16* gA = A + (size_t)(m0 + ra) * K + ca;

  int rb[2], cb[2], gn[2];
  const _Float16* gB[2];
#pragma unroll
  for (int it = 0; it < 2; ++it) {
    int idx = tid + it * 256;         // 0..511
    rb[it] = idx >> 2;                // B row 0..127
    cb[it] = (idx & 3) << 3;
    gn[it] = n0 + rb[it];
    gB[it] = B + (size_t)gn[it] * K + cb[it];
  }

  v8f acc[2][2] = {};
  const int nk = K / GTK;

  // prologue: stage K-tile 0 into buffer 0
  {
    v8h va = *(const v8h*)gA;
    *(v8h*)&As[0][ra * APITCH + ca] = va;
#pragma unroll
    for (int it = 0; it < 2; ++it) {
      v8h vb = {};
      if (gn[it] < N) vb = *(const v8h*)gB[it];
      *(v8h*)&Bs[0][rb[it] * BPITCH + cb[it]] = vb;
    }
  }
  __syncthreads();

  for (int ks = 0; ks < nk; ++ks) {
    const int cur = ks & 1;
    const int nxt = cur ^ 1;
    const bool has_next = (ks + 1) < nk;

    // issue next tile's global loads up front (latency overlaps WMMA work)
    v8h va, vb0, vb1;
    if (has_next) {
      const int ko = (ks + 1) * GTK;
      va  = *(const v8h*)(gA + ko);
      vb0 = {};
      vb1 = {};
      if (gn[0] < N) vb0 = *(const v8h*)(gB[0] + ko);
      if (gn[1] < N) vb1 = *(const v8h*)(gB[1] + ko);
      if (ks + 2 < nk) {  // L2 prefetch two tiles ahead
        const int kp = (ks + 2) * GTK;
        __builtin_prefetch(gA + kp, 0, 1);
        __builtin_prefetch(gB[0] + kp, 0, 1);
        __builtin_prefetch(gB[1] + kp, 0, 1);
      }
    }

    // compute current tile from LDS
    v16h a0 = load_frag(&As[cur][(wm * 32 +  0) * APITCH], APITCH);
    v16h a1 = load_frag(&As[cur][(wm * 32 + 16) * APITCH], APITCH);
    v16h b0 = load_frag(&Bs[cur][(wn * 32 +  0) * BPITCH], BPITCH);
    v16h b1 = load_frag(&Bs[cur][(wn * 32 + 16) * BPITCH], BPITCH);
    acc[0][0] = wmma32(a0, b0, acc[0][0]);
    acc[0][1] = wmma32(a0, b1, acc[0][1]);
    acc[1][0] = wmma32(a1, b0, acc[1][0]);
    acc[1][1] = wmma32(a1, b1, acc[1][1]);

    if (has_next) {
      *(v8h*)&As[nxt][ra * APITCH + ca] = va;
      *(v8h*)&Bs[nxt][rb[0] * BPITCH + cb[0]] = vb0;
      *(v8h*)&Bs[nxt][rb[1] * BPITCH + cb[1]] = vb1;
      __syncthreads();      // single barrier per K step
    }
  }

  const int lr = (lane >> 4) << 3;   // row base within 16x16 tile
  const int lc = lane & 15;          // col within tile
#pragma unroll
  for (int i = 0; i < 2; ++i)
#pragma unroll
    for (int j = 0; j < 2; ++j) {
      int m = m0 + wm * 32 + i * 16 + lr;
      int n = n0 + wn * 32 + j * 16 + lc;
      if (n < N) {
        float* cp = C + (size_t)m * N + n;
#pragma unroll
        for (int e = 0; e < 8; ++e) cp[(size_t)e * N] = acc[i][j][e];
      }
    }
}

// ---------------------------------------------------------------------------
// RMSNorm rows -> f16.  One 256-thread block per row.
// ---------------------------------------------------------------------------
__global__ __launch_bounds__(256)
void k_rmsnorm(const float* __restrict__ x, const float* __restrict__ w,
               _Float16* __restrict__ y, int ld_in, int D, int ld_out) {
  const int s = blockIdx.x;
  const float* row = x + (size_t)s * ld_in;
  float ss = 0.f;
  for (int i = threadIdx.x; i < D; i += 256) { float v = row[i]; ss += v * v; }
#pragma unroll
  for (int off = 16; off; off >>= 1) ss += __shfl_xor(ss, off, 32);
  __shared__ float red[8];
  if ((threadIdx.x & 31) == 0) red[threadIdx.x >> 5] = ss;
  __syncthreads();
  if (threadIdx.x == 0) {
    float v = 0.f;
    for (int i = 0; i < 8; ++i) v += red[i];
    red[0] = v;
  }
  __syncthreads();
  float inv = rsqrtf(red[0] / (float)D + 1e-6f);
  for (int i = threadIdx.x; i < D; i += 256)
    y[(size_t)s * ld_out + i] = (_Float16)(row[i] * inv * w[i]);
}

// ---------------------------------------------------------------------------
// LayerNorm(128) + rope(first 64) on the indexer keys -> f16.  128 thr/row.
// ---------------------------------------------------------------------------
__global__ __launch_bounds__(128)
void k_ln_rope_idxk(const float* __restrict__ xk, const float* __restrict__ w,
                    const float* __restrict__ b, _Float16* __restrict__ out) {
  const int s = blockIdx.x;
  const int t = threadIdx.x;           // 0..127
  float v = xk[(size_t)s * IDXD + t];
  float sum = v, sq = v * v;
#pragma unroll
  for (int off = 16; off; off >>= 1) {
    sum += __shfl_xor(sum, off, 32);
    sq  += __shfl_xor(sq,  off, 32);
  }
  __shared__ float s1[4], s2[4];
  if ((t & 31) == 0) { s1[t >> 5] = sum; s2[t >> 5] = sq; }
  __syncthreads();
  if (t == 0) {
    float a = s1[0] + s1[1] + s1[2] + s1[3];
    float c = s2[0] + s2[1] + s2[2] + s2[3];
    s1[0] = a; s2[0] = c;
  }
  __syncthreads();
  float mu  = s1[0] * (1.f / 128.f);
  float var = s2[0] * (1.f / 128.f) - mu * mu;
  float norm = (v - mu) * rsqrtf(var + 1e-6f) * w[t] + b[t];
  __shared__ float sh[128];
  sh[t] = norm;
  __syncthreads();
  float o;
  if (t < 32) {
    float a = (float)s * inv_freq(t);
    o = sh[t] * __cosf(a) - sh[t + 32] * __sinf(a);
  } else if (t < 64) {
    float a = (float)s * inv_freq(t - 32);
    o = sh[t] * __cosf(a) + sh[t - 32] * __sinf(a);
  } else {
    o = norm;
  }
  out[(size_t)s * IDXD + t] = (_Float16)o;
}

// ---------------------------------------------------------------------------
// In-place rope of a 64-dim slice at (row, base_off) of a row-major buffer.
// 8 (row) pairs per 256-thread block, 32 lanes per pair.
// ---------------------------------------------------------------------------
__global__ __launch_bounds__(256)
void k_rope64(float* __restrict__ buf, int row_stride, int base_off,
              int rows_per_pos) {
  int gid = blockIdx.x * 8 + (threadIdx.x >> 5);
  int i   = threadIdx.x & 31;
  int pos = gid / rows_per_pos;              // position id s
  float* p = buf + (size_t)gid * row_stride + base_off;
  float a = (float)pos * inv_freq(i);
  float c = __cosf(a), sn = __sinf(a);
  float x0 = p[i], x1 = p[i + 32];
  p[i]      = x0 * c - x1 * sn;
  p[i + 32] = x1 * c + x0 * sn;
}

// ---------------------------------------------------------------------------
// Fold indexer heads: qfold[s,d] = scale * sum_h idx_w[s,h] * idx_q[s,h,d]
// scale = IDX_H^-0.5 * IDX_D^-0.5
// ---------------------------------------------------------------------------
__global__ __launch_bounds__(256)
void k_fold(const float* __restrict__ idx_q, const float* __restrict__ idx_w,
            _Float16* __restrict__ qfold) {
  int idx = blockIdx.x * 256 + threadIdx.x;
  int s = idx >> 7;
  int d = idx & 127;
  const float* wr = idx_w + (size_t)s * IDXH;
  const float* qr = idx_q + (size_t)s * (IDXH * IDXD) + d;
  float acc = 0.f;
#pragma unroll
  for (int h = 0; h < IDXH; ++h) acc += wr[h] * qr[h * IDXD];
  qfold[idx] = (_Float16)(acc * 0.25f * 0.08838834764831845f);
}

// ---------------------------------------------------------------------------
// Per-row top-k threshold via bisection: tau[s] ~ 512th-largest score
// among t in [0, s].  Rows with <=512 valid entries get tau = -inf.
// ---------------------------------------------------------------------------
__global__ __launch_bounds__(256)
void k_tau(const float* __restrict__ score, float* __restrict__ tau) {
  const int s = blockIdx.x;
  const int nvalid = s + 1;
  if (nvalid <= TOPK) {
    if (threadIdx.x == 0) tau[s] = NEGBIG;
    return;
  }
  const float* row = score + (size_t)s * S_LEN;
  float mx = NEGBIG, mn = -NEGBIG;
  for (int t = threadIdx.x; t < nvalid; t += 256) {
    float v = row[t];
    mx = fmaxf(mx, v);
    mn = fminf(mn, v);
  }
#pragma unroll
  for (int off = 16; off; off >>= 1) {
    mx = fmaxf(mx, __shfl_xor(mx, off, 32));
    mn = fminf(mn, __shfl_xor(mn, off, 32));
  }
  __shared__ float smx[8], smn[8];
  __shared__ int cnt;
  if ((threadIdx.x & 31) == 0) { smx[threadIdx.x >> 5] = mx; smn[threadIdx.x >> 5] = mn; }
  __syncthreads();
  if (threadIdx.x == 0) {
    for (int i = 1; i < 8; ++i) { smx[0] = fmaxf(smx[0], smx[i]); smn[0] = fminf(smn[0], smn[i]); }
  }
  __syncthreads();
  float lo = smn[0], hi = smx[0];
  for (int it = 0; it < 32; ++it) {
    if (threadIdx.x == 0) cnt = 0;
    __syncthreads();
    float mid = 0.5f * (lo + hi);
    int c = 0;
    for (int t = threadIdx.x; t < nvalid; t += 256) c += (row[t] >= mid) ? 1 : 0;
#pragma unroll
    for (int off = 16; off; off >>= 1) c += __shfl_xor(c, off, 32);
    if ((threadIdx.x & 31) == 0) atomicAdd(&cnt, c);
    __syncthreads();
    int total = cnt;
    __syncthreads();   // protect cnt before next-iteration reset
    if (total >= TOPK) lo = mid; else hi = mid;
  }
  if (threadIdx.x == 0) tau[s] = lo;
}

// ---------------------------------------------------------------------------
// Flash attention per (head, 64-query tile).  128 threads = 4 waves; each
// wave owns 16 query rows.  Q fragments in registers; K/V^T staged to LDS
// as f16; scores + P@V via v_wmma_f32_16x16x32_f16 with online softmax.
// Masking: causal (t<=s) AND indexer top-k (idx_score[s,t] >= tau[s]).
// ---------------------------------------------------------------------------
#define AKT 64
#define KPITCH 200   // >=192, 16B-aligned rows
#define VPITCH 72    // >=64
#define PPITCH 72

__global__ __launch_bounds__(128)
void k_attn(const float* __restrict__ q,        // (S, NH, 192) roped
            const float* __restrict__ kv,       // (S, NH, 256)
            const float* __restrict__ kva,      // (S, 576) cols 512.. = roped k_pe
            const float* __restrict__ idx_score,// (S, S)
            const float* __restrict__ tau,      // (S)
            float* __restrict__ out) {          // (S, NH*128)
  __shared__ _Float16 Ks[AKT * KPITCH];
  __shared__ _Float16 Vt[D_V * VPITCH];
  __shared__ _Float16 Ps[4 * 16 * PPITCH];

  const int n    = blockIdx.y;
  const int q0   = blockIdx.x * 64;
  const int tid  = threadIdx.x;
  const int lane = tid & 31;
  const int wave = tid >> 5;
  const int lr   = (lane >> 4) << 3;
  const int lc   = lane & 15;
  const int qrow = q0 + wave * 16;

  // --- load this wave's Q fragments (16 x 192 in 6 K-steps), f32->f16 ---
  v16h qf[6];
  {
    const int r  = lane & 15;
    const int kb = (lane >> 4) << 3;
    const float* qp = q + ((size_t)(qrow + r) * NH + n) * D_QK;
#pragma unroll
    for (int f = 0; f < 6; ++f) {
      const int k0 = f * 32;
#pragma unroll
      for (int i = 0; i < 8; ++i) {
        qf[f][i]     = (_Float16)qp[k0 + kb + i];
        qf[f][8 + i] = (_Float16)qp[k0 + kb + 16 + i];
      }
    }
  }

  v8f   o[8] = {};
  float mrow[8], lrow[8];
#pragma unroll
  for (int e = 0; e < 8; ++e) { mrow[e] = NEGBIG; lrow[e] = 0.f; }

  const float sc = rsqrtf((float)D_QK);
  const int maxT = q0 + 63;

  for (int t0 = 0; t0 <= maxT; t0 += AKT) {
    // stage K tile (64 x 192): nope from kv, pe from kva
    for (int idx = tid; idx < AKT * D_QK; idx += 128) {
      int r = idx / D_QK, c = idx % D_QK;
      float v = (c < D_NOPE)
          ? kv[((size_t)(t0 + r) * NH + n) * 256 + c]
          : kva[(size_t)(t0 + r) * 576 + KV_LORA + (c - D_NOPE)];
      Ks[r * KPITCH + c] = (_Float16)v;
    }
    // stage V transposed: Vt[d][r]
    for (int idx = tid; idx < AKT * D_V; idx += 128) {
      int r = idx >> 7, d = idx & 127;
      Vt[d * VPITCH + r] =
          (_Float16)kv[((size_t)(t0 + r) * NH + n) * 256 + 128 + d];
    }
    // L2 prefetch of next tile's kv rows (global_prefetch_b8)
    if (t0 + AKT <= maxT) {
      int r = tid >> 1;                 // 0..63
      __builtin_prefetch(kv + ((size_t)(t0 + AKT + r) * NH + n) * 256 +
                             ((tid & 1) ? 128 : 0), 0, 1);
    }
    __syncthreads();

    // scores: 16 x 64 per wave
    v8f sacc[4] = {};
#pragma unroll
    for (int f = 0; f < 6; ++f)
#pragma unroll
      for (int j = 0; j < 4; ++j) {
        v16h bb = load_frag(&Ks[(j * 16) * KPITCH + f * 32], KPITCH);
        sacc[j] = wmma32(qf[f], bb, sacc[j]);
      }

    // scale + mask + row max
    float nm[8];
#pragma unroll
    for (int e = 0; e < 8; ++e) nm[e] = mrow[e];
#pragma unroll
    for (int j = 0; j < 4; ++j) {
      const int t = t0 + j * 16 + lc;
#pragma unroll
      for (int e = 0; e < 8; ++e) {
        const int s = qrow + lr + e;
        float v = sacc[j][e] * sc;
        bool ok = (t <= s) && (idx_score[(size_t)s * S_LEN + t] >= tau[s]);
        v = ok ? v : NEGBIG;
        sacc[j][e] = v;
        nm[e] = fmaxf(nm[e], v);
      }
    }
#pragma unroll
    for (int e = 0; e < 8; ++e)
#pragma unroll
      for (int off = 1; off < 16; off <<= 1)
        nm[e] = fmaxf(nm[e], __shfl_xor(nm[e], off, 32));

    float corr[8];
#pragma unroll
    for (int e = 0; e < 8; ++e) {
      corr[e] = __expf(mrow[e] - nm[e]);
      mrow[e] = nm[e];
      lrow[e] *= corr[e];
    }

    // exponentiate -> Ps (A-operand staging), accumulate row sums
    _Float16* pw = &Ps[wave * 16 * PPITCH];
    float psum[8] = {};
#pragma unroll
    for (int j = 0; j < 4; ++j)
#pragma unroll
      for (int e = 0; e < 8; ++e) {
        float v = sacc[j][e];
        float p = (v > -1e29f) ? __expf(v - mrow[e]) : 0.f;
        psum[e] += p;
        pw[(lr + e) * PPITCH + j * 16 + lc] = (_Float16)p;
      }
#pragma unroll
    for (int e = 0; e < 8; ++e) {
#pragma unroll
      for (int off = 1; off < 16; off <<= 1)
        psum[e] += __shfl_xor(psum[e], off, 32);
      lrow[e] += psum[e];
    }

    // rescale O, then O += P @ V
#pragma unroll
    for (int d = 0; d < 8; ++d)
#pragma unroll
      for (int e = 0; e < 8; ++e) o[d][e] *= corr[e];

    v16h pa0 = load_frag(pw, PPITCH);
    v16h pa1 = load_frag(pw + 32, PPITCH);
#pragma unroll
    for (int d = 0; d < 8; ++d) {
      v16h b0 = load_frag(&Vt[(d * 16) * VPITCH + 0],  VPITCH);
      v16h b1 = load_frag(&Vt[(d * 16) * VPITCH + 32], VPITCH);
      o[d] = wmma32(pa0, b0, o[d]);
      o[d] = wmma32(pa1, b1, o[d]);
    }
    __syncthreads();
  }

#pragma unroll
  for (int d = 0; d < 8; ++d) {
    const int col = n * D_V + d * 16 + lc;
#pragma unroll
    for (int e = 0; e < 8; ++e) {
      const int s = qrow + lr + e;
      out[(size_t)s * (NH * D_V) + col] = o[d][e] / fmaxf(lrow[e], 1e-20f);
    }
  }
}

// ---------------------------------------------------------------------------
// host launch
// ---------------------------------------------------------------------------
extern "C" void kernel_launch(void* const* d_in, const int* in_sizes, int n_in,
                              void* d_out, int out_size, void* d_ws, size_t ws_size,
                              hipStream_t stream) {
  const float* x        = (const float*)d_in[0];
  // d_in[1] position_ids == arange(S): positions derived from row index
  const float* q_a_w    = (const float*)d_in[2];
  const float* q_a_ln_w = (const float*)d_in[3];
  const float* q_b_w    = (const float*)d_in[4];
  const float* kv_a_w   = (const float*)d_in[5];
  const float* kv_a_ln_w= (const float*)d_in[6];
  const float* kv_b_w   = (const float*)d_in[7];
  const float* o_w      = (const float*)d_in[8];
  const float* wq_b_w   = (const float*)d_in[9];
  const float* wk_w     = (const float*)d_in[10];
  const float* k_norm_w = (const float*)d_in[11];
  const float* k_norm_b = (const float*)d_in[12];
  const float* wproj_w  = (const float*)d_in[13];

  char* bump = (char*)d_ws;
  auto alloc = [&](size_t bytes) -> void* {
    void* r = (void*)bump;
    bump += (bytes + 255) & ~(size_t)255;
    return r;
  };

  _Float16* x_h      = (_Float16*)alloc((size_t)S_LEN * HID * 2);
  _Float16* wqa_h    = (_Float16*)alloc((size_t)Q_LORA * HID * 2);
  _Float16* wqb_h    = (_Float16*)alloc((size_t)NH * D_QK * Q_LORA * 2);
  _Float16* wkva_h   = (_Float16*)alloc((size_t)(KV_LORA + D_ROPE) * HID * 2);
  _Float16* wkvb_h   = (_Float16*)alloc((size_t)NH * 256 * KV_LORA * 2);
  _Float16* wo_h     = (_Float16*)alloc((size_t)HID * NH * D_V * 2);
  _Float16* wwqb_h   = (_Float16*)alloc((size_t)IDXH * IDXD * Q_LORA * 2);
  _Float16* wwk_h    = (_Float16*)alloc((size_t)IDXD * HID * 2);
  _Float16* wwp_h    = (_Float16*)alloc((size_t)IDXH * HID * 2);
  float*    qa_f     = (float*)   alloc((size_t)S_LEN * Q_LORA * 4);
  _Float16* qr_h     = (_Float16*)alloc((size_t)S_LEN * Q_LORA * 2);
  float*    q_f      = (float*)   alloc((size_t)S_LEN * NH * D_QK * 4);
  float*    kva_f    = (float*)   alloc((size_t)S_LEN * (KV_LORA + D_ROPE) * 4);
  _Float16* comp_h   = (_Float16*)alloc((size_t)S_LEN * KV_LORA * 2);
  float*    kv_f     = (float*)   alloc((size_t)S_LEN * NH * 256 * 4);
  float*    idxq_f   = (float*)   alloc((size_t)S_LEN * IDXH * IDXD * 4);
  float*    idxkraw_f= (float*)   alloc((size_t)S_LEN * IDXD * 4);
  _Float16* idxk_h   = (_Float16*)alloc((size_t)S_LEN * IDXD * 2);
  float*    idxw_f   = (float*)   alloc((size_t)S_LEN * IDXH * 4);
  _Float16* qfold_h  = (_Float16*)alloc((size_t)S_LEN * IDXD * 2);
  float*    score_f  = (float*)   alloc((size_t)S_LEN * S_LEN * 4);
  float*    tau_f    = (float*)   alloc((size_t)S_LEN * 4);
  float*    ao_f     = (float*)   alloc((size_t)S_LEN * NH * D_V * 4);
  _Float16* ao_h     = (_Float16*)alloc((size_t)S_LEN * NH * D_V * 2);

  auto cvt = [&](const float* s, _Float16* d, size_t n) {
    k_cvt<<<dim3((unsigned)((n + 255) / 256)), 256, 0, stream>>>(s, d, (int)n);
  };
  auto gemm = [&](const _Float16* A, const _Float16* B, float* C,
                  int M, int N, int K) {
    dim3 g((N + GTN - 1) / GTN, M / GTM);
    k_gemm_nt<<<g, 256, 0, stream>>>(A, B, C, M, N, K);
  };

  // 1) f32 -> f16 conversions (activations + weights)
  cvt(x,      x_h,    (size_t)S_LEN * HID);
  cvt(q_a_w,  wqa_h,  (size_t)Q_LORA * HID);
  cvt(q_b_w,  wqb_h,  (size_t)NH * D_QK * Q_LORA);
  cvt(kv_a_w, wkva_h, (size_t)(KV_LORA + D_ROPE) * HID);
  cvt(kv_b_w, wkvb_h, (size_t)NH * 256 * KV_LORA);
  cvt(o_w,    wo_h,   (size_t)HID * NH * D_V);
  cvt(wq_b_w, wwqb_h, (size_t)IDXH * IDXD * Q_LORA);
  cvt(wk_w,   wwk_h,  (size_t)IDXD * HID);
  cvt(wproj_w,wwp_h,  (size_t)IDXH * HID);

  // 2) q_a + rmsnorm
  gemm(x_h, wqa_h, qa_f, S_LEN, Q_LORA, HID);
  k_rmsnorm<<<S_LEN, 256, 0, stream>>>(qa_f, q_a_ln_w, qr_h, Q_LORA, Q_LORA, Q_LORA);

  // 3) q projection + rope on pe slice
  gemm(qr_h, wqb_h, q_f, S_LEN, NH * D_QK, Q_LORA);
  k_rope64<<<(S_LEN * NH) / 8, 256, 0, stream>>>(q_f, D_QK, D_NOPE, NH);

  // 4) kv_a + rmsnorm(512) + rope on k_pe
  gemm(x_h, wkva_h, kva_f, S_LEN, KV_LORA + D_ROPE, HID);
  k_rmsnorm<<<S_LEN, 256, 0, stream>>>(kva_f, kv_a_ln_w, comp_h,
                                       KV_LORA + D_ROPE, KV_LORA, KV_LORA);
  k_rope64<<<S_LEN / 8, 256, 0, stream>>>(kva_f, KV_LORA + D_ROPE, KV_LORA, 1);

  // 5) kv expansion
  gemm(comp_h, wkvb_h, kv_f, S_LEN, NH * 256, KV_LORA);

  // 6) indexer queries + rope
  gemm(qr_h, wwqb_h, idxq_f, S_LEN, IDXH * IDXD, Q_LORA);
  k_rope64<<<(S_LEN * IDXH) / 8, 256, 0, stream>>>(idxq_f, IDXD, 0, IDXH);

  // 7) indexer keys: gemm + layernorm + rope -> f16
  gemm(x_h, wwk_h, idxkraw_f, S_LEN, IDXD, HID);
  k_ln_rope_idxk<<<S_LEN, 128, 0, stream>>>(idxkraw_f, k_norm_w, k_norm_b, idxk_h);

  // 8) indexer head weights, fold heads into one 128-dim query
  gemm(x_h, wwp_h, idxw_f, S_LEN, IDXH, HID);
  k_fold<<<(S_LEN * IDXD) / 256, 256, 0, stream>>>(idxq_f, idxw_f, qfold_h);

  // 9) indexer scores (S x S) + per-row top-k threshold
  gemm(qfold_h, idxk_h, score_f, S_LEN, S_LEN, IDXD);
  k_tau<<<S_LEN, 256, 0, stream>>>(score_f, tau_f);

  // 10) sparse flash attention
  k_attn<<<dim3(S_LEN / 64, NH), 128, 0, stream>>>(q_f, kv_f, kva_f,
                                                   score_f, tau_f, ao_f);

  // 11) output projection -> d_out (f32)
  cvt(ao_f, ao_h, (size_t)S_LEN * NH * D_V);
  gemm(ao_h, wo_h, (float*)d_out, S_LEN, HID, NH * D_V);
}
